// OpenRAGGateAdapterRoE_3908420239749
// MI455X (gfx1250) — matrix operand order repaired and checked by reading, not compiled, not run on hardware
//
#include <hip/hip_runtime.h>
#include <math.h>

// Problem constants (match reference)
namespace {
constexpr int kH = 4096;          // hidden
constexpr int kE = 8;             // experts
constexpr int kA = 64;            // adapter rank
constexpr int kN = 4 * 2048;      // B*S = 8192 tokens
constexpr int kCH = 128;          // TDM K-chunk (elements) = 4 WMMA K-steps
constexpr float kScale = 2.0f;

typedef __bf16 v16bf __attribute__((ext_vector_type(16)));
typedef __bf16 v2bf __attribute__((ext_vector_type(2)));
typedef float v8f __attribute__((ext_vector_type(8)));
typedef unsigned short v16u __attribute__((ext_vector_type(16)));
typedef unsigned int u32x2 __attribute__((ext_vector_type(2)));
typedef unsigned int u32x4 __attribute__((ext_vector_type(4)));
typedef int i32x4 __attribute__((ext_vector_type(4)));
typedef int i32x8 __attribute__((ext_vector_type(8)));

__device__ __forceinline__ unsigned short f32_to_bf16(float f) {
  unsigned u = __builtin_bit_cast(unsigned, f);
  u += 0x7FFFu + ((u >> 16) & 1u);   // round-to-nearest-even
  return (unsigned short)(u >> 16);
}

__device__ __forceinline__ v16bf pack16(const float* g) {
#if __has_builtin(__builtin_amdgcn_cvt_pk_bf16_f32)
  v16bf v;
#pragma unroll
  for (int i = 0; i < 8; ++i) {
    v2bf p = __builtin_amdgcn_cvt_pk_bf16_f32(g[2 * i], g[2 * i + 1]);
    v[2 * i] = p[0];
    v[2 * i + 1] = p[1];
  }
  return v;
#else
  v16u u;
#pragma unroll
  for (int i = 0; i < 16; ++i) u[i] = f32_to_bf16(g[i]);
  return __builtin_bit_cast(v16bf, u);
#endif
}

struct BFrag { u32x4 lo, hi; };     // 32B = 16 bf16 = one per-lane fragment
// Fragment from two 16B groups (works for global b128 and LDS ds_load_b128).
__device__ __forceinline__ v16bf load_frag2(const unsigned short* p0,
                                            const unsigned short* p1) {
  BFrag b;
  b.lo = *(const u32x4*)(p0);
  b.hi = *(const u32x4*)(p1);
  return __builtin_bit_cast(v16bf, b);
}

// -------- Tensor Data Mover: gather 16 bf16 rows x kCH elems into LDS ------
// D# per CDNA5 ISA ch.8: group0 = control/lds/global addr, group1 = dims,
// groups 2/3 = sixteen packed 16-bit row indices (gather mode).
__device__ __forceinline__ void tdm_gather16(unsigned lds_addr,
                                             const unsigned short* gbase,
                                             const unsigned* idxw) {
  u32x4 g0;
  g0[0] = 0x80000001u;                       // count=1 | gather_mode | 16b idx
  g0[1] = lds_addr;                          // LDS byte address
  unsigned long long ga = (unsigned long long)(uintptr_t)gbase;
  g0[2] = (unsigned)ga;                      // global addr [95:64]
  g0[3] = (unsigned)((ga >> 32) & 0x01FFFFFFull) | (2u << 30);  // [120:96]|type=2
  i32x8 g1;
  g1[0] = 0x00010000;                        // data_size=1 (2 bytes, bf16)
  g1[1] = (int)((unsigned)(kH & 0xFFFF) << 16);   // tensor_dim0 = 4096 (lo16)
  g1[2] = (int)((unsigned)(kN & 0xFFFF) << 16);   // tensor_dim1 = 8192 (lo16)
  g1[3] = (int)((unsigned)kCH << 16);             // tile_dim0 = 128
  g1[4] = 16;                                // tile_dim1 = #indices = 16
  g1[5] = kH;                                // tensor_dim0_stride = 4096
  g1[6] = 0;
  g1[7] = 0;
  i32x4 g2, g3;
  g2[0] = (int)idxw[0]; g2[1] = (int)idxw[1];
  g2[2] = (int)idxw[2]; g2[3] = (int)idxw[3];
  g3[0] = (int)idxw[4]; g3[1] = (int)idxw[5];
  g3[2] = (int)idxw[6]; g3[3] = (int)idxw[7];
#if defined(__clang_major__) && __clang_major__ >= 23
  i32x8 gz = {};
  __builtin_amdgcn_tensor_load_to_lds(g0, g1, g2, g3, gz, 0);
#else
  __builtin_amdgcn_tensor_load_to_lds(g0, g1, g2, g3, 0);
#endif
}
} // namespace

// ---------------------------------------------------------------------------
// Kernel 1: out = o  (float4 streaming copy) + zero per-expert counters
// ---------------------------------------------------------------------------
__global__ __launch_bounds__(256) void roe_init_copy(
    const float* __restrict__ o, float* __restrict__ out,
    int* __restrict__ counts) {
  if (blockIdx.x == 0 && threadIdx.x < kE) counts[threadIdx.x] = 0;
  const size_t n4 = (size_t)kN * kH / 4;
  size_t i = (size_t)blockIdx.x * blockDim.x + threadIdx.x;
  const size_t stride = (size_t)gridDim.x * blockDim.x;
  const float4* src = (const float4*)o;
  float4* dst = (float4*)out;
  for (size_t k = i; k < n4; k += stride) dst[k] = src[k];
}

// ---------------------------------------------------------------------------
// Kernel 2: streaming f32 -> bf16 conversion of x (keeps hot loop cvt-free;
// also warms L2 so the per-expert TDM gathers are mostly L2-served).
// ---------------------------------------------------------------------------
__global__ __launch_bounds__(256) void roe_cvt_x(
    const float* __restrict__ x, unsigned short* __restrict__ xb) {
  const size_t n4 = (size_t)kN * kH / 4;
  size_t i = (size_t)blockIdx.x * blockDim.x + threadIdx.x;
  const size_t stride = (size_t)gridDim.x * blockDim.x;
  for (size_t k = i; k < n4; k += stride) {
    float4 v = ((const float4*)x)[k];
    u32x2 p;
    p[0] = (unsigned)f32_to_bf16(v.x) | ((unsigned)f32_to_bf16(v.y) << 16);
    p[1] = (unsigned)f32_to_bf16(v.z) | ((unsigned)f32_to_bf16(v.w) << 16);
    *(u32x2*)(xb + k * 4) = p;
  }
}

// ---------------------------------------------------------------------------
// Kernels 3a/3b: pre-swizzle adapter weights into bf16 WMMA B-fragment order.
// Layout: [e][frag][lane][16 elems] -> each lane loads 32 contiguous bytes,
// each wave 1KB contiguous, zero conversion ALU in the GEMM hot loops.
// ---------------------------------------------------------------------------
__global__ __launch_bounds__(256) void roe_swz_wd(
    const float* __restrict__ Wd, unsigned short* __restrict__ wdsw) {
  int idx = blockIdx.x * 256 + threadIdx.x;   // kE*512*512 = 2,097,152
  int i = idx & 15;
  int lane = (idx >> 4) & 31;
  int f = (idx >> 9) & 511;                   // frag = kstep*4 + t
  int e = idx >> 18;
  int kstep = f >> 2, t = f & 3;
  int row = t * 16 + (lane & 15);             // adapter row (0..63)
  int col = kstep * 32 + (lane >> 4) * 8 + (i & 7) + (i >> 3) * 16;
  wdsw[idx] = f32_to_bf16(Wd[((size_t)e * kA + row) * kH + col]);
}

__global__ __launch_bounds__(256) void roe_swz_wu(
    const float* __restrict__ Wu, unsigned short* __restrict__ wusw) {
  int idx = blockIdx.x * 256 + threadIdx.x;   // kE*512*512 = 2,097,152
  int i = idx & 15;
  int lane = (idx >> 4) & 31;
  int f = (idx >> 9) & 511;                   // frag = (jb/16)*2 + s
  int e = idx >> 18;
  int s = f & 1, jt = f >> 1;
  int j = jt * 16 + (lane & 15);              // output column in H
  int a = s * 32 + (lane >> 4) * 8 + (i & 7) + (i >> 3) * 16;      // adapter K index
  wusw[idx] = f32_to_bf16(Wu[((size_t)e * kH + j) * kA + a]);
}

// ---------------------------------------------------------------------------
// Kernel 4: router logits (f32 exact) + top-2 + softmax + expert scatter.
// ---------------------------------------------------------------------------
__global__ __launch_bounds__(256) void roe_router_top2(
    const float* __restrict__ r, const float* __restrict__ Wr,
    float* __restrict__ logits, int* __restrict__ counts,
    int* __restrict__ ltok, float* __restrict__ lw) {
  const int lane = threadIdx.x & 31;
  const int wave = threadIdx.x >> 5;
  const int n = blockIdx.x * 8 + wave;
  const float* rrow = r + (size_t)n * kH;

  float acc[kE];
#pragma unroll
  for (int e = 0; e < kE; ++e) acc[e] = 0.f;

  for (int c = lane * 4; c < kH; c += 32 * 4) {
    float4 rv = *(const float4*)(rrow + c);
#pragma unroll
    for (int e = 0; e < kE; ++e) {
      float4 wv = *(const float4*)(Wr + (size_t)e * kH + c);
      acc[e] = fmaf(rv.x, wv.x, acc[e]);
      acc[e] = fmaf(rv.y, wv.y, acc[e]);
      acc[e] = fmaf(rv.z, wv.z, acc[e]);
      acc[e] = fmaf(rv.w, wv.w, acc[e]);
    }
  }
#pragma unroll
  for (int e = 0; e < kE; ++e) {
#pragma unroll
    for (int off = 16; off > 0; off >>= 1)
      acc[e] += __shfl_xor(acc[e], off, 32);
  }

  if (lane == 0) {
    float* lrow = logits + (size_t)n * kE;
#pragma unroll
    for (int e = 0; e < kE; ++e) lrow[e] = acc[e];
    int b0 = 0; float v0 = acc[0];
#pragma unroll
    for (int e = 1; e < kE; ++e) if (acc[e] > v0) { v0 = acc[e]; b0 = e; }
    int b1 = (b0 == 0) ? 1 : 0; float v1 = -3.4e38f;
#pragma unroll
    for (int e = 0; e < kE; ++e)
      if (e != b0 && acc[e] > v1) { v1 = acc[e]; b1 = e; }
    float t = expf(v1 - v0);
    float inv = 1.f / (1.f + t);
    int pos0 = atomicAdd(&counts[b0], 1);
    ltok[b0 * kN + pos0] = n; lw[b0 * kN + pos0] = inv;
    int pos1 = atomicAdd(&counts[b1], 1);
    ltok[b1 * kN + pos1] = n; lw[b1 * kN + pos1] = t * inv;
  }
}

// ---------------------------------------------------------------------------
// Kernel 5: fused expert adapter.
// TDM gather-mode double-buffered bf16 x staging -> bf16 WMMA down-proj ->
// exact GELU -> LDS transpose -> bf16 WMMA up-proj -> f32 atomic combine.
// Grid (N/16, E), one wave32 per block, uniform early exit.
// ---------------------------------------------------------------------------
__global__ __launch_bounds__(32) void roe_adapter_fused(
    const unsigned short* __restrict__ xb, const unsigned short* __restrict__ wdsw,
    const unsigned short* __restrict__ wusw, const int* __restrict__ counts,
    const int* __restrict__ ltok, const float* __restrict__ lw,
    float* __restrict__ out) {
  const int e = blockIdx.y;
  const int cnt = counts[e];                 // uniform scalar load
  const int base = blockIdx.x * 16;
  if (base >= cnt) return;                   // uniform exit, EXEC stays full

  const int lane = threadIdx.x;              // 0..31
  const int nsub = lane & 15;
  const int hi = lane >> 4;
  const int mofs = hi * 8;                   // C-tile row offset for this half
  const int koffA = hi * 8;                  // A-layout K groups per half

  const int myrow = base + nsub;
  int tokA = 0; float wA = 0.f;
  if (myrow < cnt) {
    tokA = ltok[e * kN + myrow];
    wA = kScale * lw[e * kN + myrow];        // fold SCALING into combine weight
  }

  // Pack the 16 gathered row ids into uniform words for the TDM descriptor.
  unsigned idxw[8];
#pragma unroll
  for (int q = 0; q < 8; ++q) {
    unsigned lo16 = (unsigned)__builtin_amdgcn_readlane(tokA, 2 * q) & 0xFFFFu;
    unsigned hi16 = (unsigned)__builtin_amdgcn_readlane(tokA, 2 * q + 1) & 0xFFFFu;
    idxw[q] = lo16 | (hi16 << 16);
  }

  __shared__ unsigned short lx[2][16][kCH];  // 8KB double-buffered bf16 chunks
  __shared__ float hlds[16][72];             // GELU(h) transpose staging

  const unsigned lds0 = (unsigned)(uintptr_t)&lx[0][0][0];
  const unsigned lds1 = (unsigned)(uintptr_t)&lx[1][0][0];

  // ---- down projection: h[16 x 64] = gather(xb) * Wd^T, K = 4096 ----
  v8f accd[4] = {};
  constexpr int NC = kH / kCH;               // 32 chunks
  tdm_gather16(lds0, xb, idxw);              // prefetch chunk 0
  for (int c = 0; c < NC; ++c) {
    const int cur = c & 1;
    if (c + 1 < NC) {
      asm volatile("s_wait_dscnt 0x0" ::: "memory");  // buffer reuse fence
      tdm_gather16(cur ? lds0 : lds1, xb + (size_t)(c + 1) * kCH, idxw);
      __builtin_amdgcn_s_wait_tensorcnt(1);  // chunk c landed
    } else {
      __builtin_amdgcn_s_wait_tensorcnt(0);
    }
    asm volatile("" ::: "memory");
#pragma unroll
    for (int ks = 0; ks < 4; ++ks) {         // 4 K-steps of 32 per chunk
      const int kl = ks * 32;
      // A fragment: two ds_load_b128 straight into WMMA sources (no cvt)
      v16bf afrag = load_frag2(&lx[cur][nsub][kl + koffA],
                               &lx[cur][nsub][kl + koffA + 16]);
      const size_t fbase = ((size_t)e * 512 + (size_t)(c * 4 + ks) * 4) * 512;
#pragma unroll
      for (int t = 0; t < 4; ++t) {
        const unsigned short* p = wdsw + fbase + (size_t)t * 512 + lane * 16;
        v16bf bfrag = load_frag2(p, p + 8);
        accd[t] = __builtin_amdgcn_wmma_f32_16x16x32_bf16(
            false, afrag, false, bfrag, (short)0, accd[t], false, false);
      }
    }
  }

  // ---- exact GELU, transpose h through LDS into A-fragment layout ----
#pragma unroll
  for (int t = 0; t < 4; ++t)
#pragma unroll
    for (int j = 0; j < 8; ++j) {
      float v = accd[t][j];
      hlds[j + mofs][t * 16 + nsub] =
          0.5f * v * (1.f + erff(v * 0.70710678118654752f));
    }
  __syncthreads();

  v16bf ua[2];
#pragma unroll
  for (int s = 0; s < 2; ++s) {
    float ta[16];
#pragma unroll
    for (int i = 0; i < 8; ++i) {
      ta[i] = hlds[nsub][s * 32 + koffA + i];
      ta[8 + i] = hlds[nsub][s * 32 + koffA + 16 + i];
    }
    ua[s] = pack16(ta);
  }

  // Broadcast per-output-row token id / weight to all lanes.
  int tokrow[8]; float wrow[8];
#pragma unroll
  for (int j = 0; j < 8; ++j) {
    tokrow[j] = __shfl(tokA, j + mofs, 32);
    wrow[j] = __shfl(wA, j + mofs, 32);
  }

  // ---- up projection: y[16 x H] = h * Wu^T, scaled + atomic combine ----
#pragma unroll 2
  for (int jb = 0; jb < kH; jb += 16) {
    v8f acc = {};
    const size_t fbase = ((size_t)e * 512 + (size_t)(jb >> 4) * 2) * 512;
#pragma unroll
    for (int s = 0; s < 2; ++s) {
      const unsigned short* p = wusw + fbase + (size_t)s * 512 + lane * 16;
      v16bf bfrag = load_frag2(p, p + 8);
      acc = __builtin_amdgcn_wmma_f32_16x16x32_bf16(
          false, ua[s], false, bfrag, (short)0, acc, false, false);
    }
#pragma unroll
    for (int j = 0; j < 8; ++j) {
      unsafeAtomicAdd(out + (size_t)tokrow[j] * kH + jb + nsub,
                      acc[j] * wrow[j]);
    }
  }
}

// ---------------------------------------------------------------------------
// Host launch
// ---------------------------------------------------------------------------
extern "C" void kernel_launch(void* const* d_in, const int* in_sizes, int n_in,
                              void* d_out, int out_size, void* d_ws,
                              size_t ws_size, hipStream_t stream) {
  (void)in_sizes; (void)n_in; (void)out_size; (void)ws_size;
  const float* x  = (const float*)d_in[0];   // input_hidden_states [B,S,H]
  const float* o  = (const float*)d_in[1];   // output_hidden_states [B,S,H]
  const float* r  = (const float*)d_in[2];   // router_hidden_states [B,S,H]
  const float* Wr = (const float*)d_in[3];   // [E,H]
  const float* Wd = (const float*)d_in[4];   // [E,A,H]
  const float* Wu = (const float*)d_in[5];   // [E,H,A]

  float* out = (float*)d_out;                // [B,S,H] then logits [N,E]
  float* logits = out + (size_t)kN * kH;

  char* ws = (char*)d_ws;
  int* counts = (int*)ws;                                   // [0,256)
  int* ltok = (int*)(ws + 256);                             // [E][N] ints
  float* lw = (float*)(ws + 256 + sizeof(int) * (size_t)kE * kN);
  unsigned short* wdsw = (unsigned short*)(ws + 524544);    // 4.19 MB
  unsigned short* wusw = wdsw + (size_t)kE * 512 * 512;     // 4.19 MB
  unsigned short* xb = wusw + (size_t)kE * 512 * 512;       // 67.1 MB bf16 x

  roe_init_copy<<<4096, 256, 0, stream>>>(o, out, counts);
  roe_cvt_x<<<8192, 256, 0, stream>>>(x, xb);
  roe_swz_wd<<<8192, 256, 0, stream>>>(Wd, wdsw);
  roe_swz_wu<<<8192, 256, 0, stream>>>(Wu, wusw);
  roe_router_top2<<<kN / 8, 256, 0, stream>>>(r, Wr, logits, counts, ltok, lw);
  roe_adapter_fused<<<dim3(kN / 16, kE), 32, 0, stream>>>(
      xb, wdsw, wusw, counts, ltok, lw, out);
}
